// SpectralCorrector_62345745268952
// MI455X (gfx1250) — matrix operand order repaired and checked by reading, not compiled, not run on hardware
//
#include <hip/hip_runtime.h>
#include <hip/hip_bf16.h>

// ---------------------------------------------------------------------------
// GCN layer for MI455X (gfx1250, wave32, WMMA):
//   agg = segment_sum(edge_weight * x[src], dst)      (SpMM, L2-resident)
//   h   = relu(concat(x,agg) @ W1 + b1)               (WMMA bf16 GEMM)
//   h   = relu(h @ W2 + b2)                           (WMMA bf16 GEMM)
//   out = h @ W3 + b3                                 (WMMA bf16 GEMM, f32 out)
// GEMMs use double-buffered LDS tiles filled with CDNA5 async global->LDS
// copies (GLOBAL_LOAD_ASYNC_TO_LDS_B128, ASYNCcnt) when available.
// ---------------------------------------------------------------------------

typedef __bf16 bf16;
typedef __attribute__((ext_vector_type(16))) __bf16 v16bf;
typedef __attribute__((ext_vector_type(8)))  float  v8f;

#define NNODES  50000
#define NEDGES  1600000
#define IN_DIM  512
#define CAT_DIM 1024
#define H1_DIM  2048
#define H2_DIM  2048
#define OUT_DIM 512

#if __has_builtin(__builtin_amdgcn_global_load_async_to_lds_b128) && \
    __has_builtin(__builtin_amdgcn_s_wait_asynccnt)
#define USE_ASYNC_LDS 1
#else
#define USE_ASYNC_LDS 0
#endif

#if USE_ASYNC_LDS
// Exact parameter types per the clang diagnostic: non-const GNU-vector int4
// pointers; AS1 (printed as __device__) for global, AS3 for LDS.
typedef int v4i_gnu __attribute__((vector_size(16)));
typedef __attribute__((address_space(1))) v4i_gnu* as1_v4i;
typedef __attribute__((address_space(3))) v4i_gnu* as3_v4i;
static __device__ __forceinline__ void async_copy16(const bf16* g, bf16* l) {
  // GLOBAL_LOAD_ASYNC_TO_LDS_B128: per-lane 16 B, tracked with ASYNCcnt.
  __builtin_amdgcn_global_load_async_to_lds_b128(
      (as1_v4i)g, (as3_v4i)l, /*imm offset*/0, /*cpol*/0);
}
#endif

// ---------------------------------------------------------------------------
// SpMM: one 128-thread block per edge; float4 gather of x[src], scale by w_e,
// f32 scatter-atomics into agg[dst]. x (102 MB) and agg (102 MB) both fit in
// the 192 MB L2, so this is an L2-atomic-bandwidth problem, small vs. GEMM.
// ---------------------------------------------------------------------------
__global__ __launch_bounds__(128) void spmm_scatter(
    const float* __restrict__ x, const float* __restrict__ ew,
    const int* __restrict__ src, const int* __restrict__ dst,
    float* __restrict__ agg) {
  int e = blockIdx.x;
  int t = threadIdx.x;                       // 0..127, 4 floats each = 512
  int s = src[e];
  int d = dst[e];
  float w = ew[e];
  float4 v = ((const float4*)(x + (size_t)s * IN_DIM))[t];
  float* o = agg + (size_t)d * IN_DIM + t * 4;
  atomicAdd(o + 0, w * v.x);
  atomicAdd(o + 1, w * v.y);
  atomicAdd(o + 2, w * v.z);
  atomicAdd(o + 3, w * v.w);
}

// ---------------------------------------------------------------------------
// W (K x N, f32, row-major)  ->  Wt (N x K, bf16, row-major).
// ---------------------------------------------------------------------------
__global__ void transpose_w_bf16(const float* __restrict__ W,
                                 bf16* __restrict__ Wt, int K, int Ncols) {
  size_t idx = (size_t)blockIdx.x * 256 + threadIdx.x;
  if (idx >= (size_t)K * Ncols) return;
  int k = (int)(idx / Ncols);
  int n = (int)(idx % Ncols);
  Wt[(size_t)n * K + k] = (bf16)W[idx];
}

// concat(x, agg) -> bf16 (N x 1024)
__global__ void build_hcat(const float* __restrict__ x,
                           const float* __restrict__ agg,
                           bf16* __restrict__ hc) {
  size_t idx = (size_t)blockIdx.x * 256 + threadIdx.x;
  if (idx >= (size_t)NNODES * CAT_DIM) return;
  size_t i = idx >> 10;
  int    c = (int)(idx & 1023);
  float  v = (c < IN_DIM) ? x[i * IN_DIM + c] : agg[i * IN_DIM + (c - IN_DIM)];
  hc[idx] = (bf16)v;
}

// ---------------------------------------------------------------------------
// GEMM: C[M,N] = act(A[M,K] @ B[K,N] + bias), B given transposed (Bt = N x K).
// Block tile 128x128, K-step 32, 8 waves: wave grid 4(M) x 2(N), each wave
// owns a 32x64 sub-tile = 2x4 v_wmma_f32_16x16x32_bf16 accumulators.
// Double-buffered LDS (40-half row pitch for bank spread); next tile copied
// asynchronously (or register-prefetched) while the current tile is computed.
// A-tile row indices are clamped to M-1 (only the last of 391 row tiles is
// ragged) so staging is branch-free and EXEC stays all-ones for WMMA.
// ---------------------------------------------------------------------------
template <bool RELU, typename OutT>
__global__ __launch_bounds__(256) void gemm_wmma_bf16(
    const bf16* __restrict__ A,   // M x K
    const bf16* __restrict__ Bt,  // N x K (transposed weights)
    const float* __restrict__ bias,
    OutT* __restrict__ C,         // M x N
    int M, int K, int Ncols) {
  constexpr int BM = 128, BN = 128, BK = 32, BKP = 40;
  __shared__ bf16 As[2][BM * BKP];
  __shared__ bf16 Bs[2][BN * BKP];

  const int tid  = threadIdx.x;
  const int lane = tid & 31;       // wave32
  const int wid  = tid >> 5;       // 0..7
  const int wm   = wid & 3;        // wave row   (4 x 32 = 128)
  const int wn   = wid >> 2;       // wave col   (2 x 64 = 128)
  const int tileM = blockIdx.y * BM;
  const int tileN = blockIdx.x * BN;
  const int ml = lane & 15;        // matrix row/col within 16
  const int kg = lane >> 4;        // lane-group selects K half

  // Staging: each thread owns chunks c=tid and c=tid+256 of each 512-chunk
  // tile (chunk = 16 B = 8 bf16; row = c>>2, k-chunk = c&3).
  const int r0 = tid >> 2, kc = tid & 3;
  const int r1 = r0 + 64;
  const int ga0 = min(tileM + r0, M - 1);     // clamped: branch-free staging
  const int ga1 = min(tileM + r1, M - 1);
  const bf16* gA0 = A  + (size_t)ga0 * K + kc * 8;
  const bf16* gA1 = A  + (size_t)ga1 * K + kc * 8;
  const bf16* gB0 = Bt + (size_t)(tileN + r0) * K + kc * 8;
  const bf16* gB1 = Bt + (size_t)(tileN + r1) * K + kc * 8;
  const int lofs0 = r0 * BKP + kc * 8;
  const int lofs1 = r1 * BKP + kc * 8;

  v8f acc[2][4];
  const v8f vzero = {0.f, 0.f, 0.f, 0.f, 0.f, 0.f, 0.f, 0.f};
#pragma unroll
  for (int i = 0; i < 2; ++i)
#pragma unroll
    for (int j = 0; j < 4; ++j) acc[i][j] = vzero;

  union Frag { v16bf v; uint4 q[2]; };

  // one K-step of WMMA from LDS buffer b
  auto compute = [&](int b) {
    Frag a[2], bb[4];
#pragma unroll
    for (int i = 0; i < 2; ++i) {
      // A 16x32 bf16: lanes0-15 v0..3=K[kg*8..+7], v4..7=K[16+kg*8..+7]
      const bf16* p = &As[b][(wm * 32 + i * 16 + ml) * BKP];
      a[i].q[0] = *(const uint4*)(p + kg * 8);
      a[i].q[1] = *(const uint4*)(p + 16 + kg * 8);
    }
#pragma unroll
    for (int j = 0; j < 4; ++j) {
      // B 32x16 bf16: lanes0-15 v0..3=K[kg*8..+7], v4..7=K[16+kg*8..+7]
      const bf16* p = &Bs[b][(wn * 64 + j * 16 + ml) * BKP];
      bb[j].q[0] = *(const uint4*)(p + kg * 8);
      bb[j].q[1] = *(const uint4*)(p + 16 + kg * 8);
    }
#pragma unroll
    for (int i = 0; i < 2; ++i)
#pragma unroll
      for (int j = 0; j < 4; ++j)
        acc[i][j] = __builtin_amdgcn_wmma_f32_16x16x32_bf16(
            false, a[i].v, false, bb[j].v, (short)0, acc[i][j], false, false);
  };

  const int nk = K / BK;

#if USE_ASYNC_LDS
  // ---- async double-buffered pipeline --------------------------------------
  auto issue_tile = [&](int k0, int b) {
    async_copy16(gA0 + k0, &As[b][lofs0]);
    async_copy16(gA1 + k0, &As[b][lofs1]);
    async_copy16(gB0 + k0, &Bs[b][lofs0]);
    async_copy16(gB1 + k0, &Bs[b][lofs1]);
  };
  issue_tile(0, 0);
  for (int kt = 0; kt < nk - 1; ++kt) {
    issue_tile((kt + 1) * BK, (kt + 1) & 1);   // prefetch next tile
    // 4 just-issued outstanding; <=4 means tile kt's 4 copies landed
    __builtin_amdgcn_s_wait_asynccnt(4);
    __syncthreads();                           // all waves' copies visible
    compute(kt & 1);                           // overlaps with in-flight copies
    __syncthreads();                           // done reading before next issue
  }
  __builtin_amdgcn_s_wait_asynccnt(0);
  __syncthreads();
  compute((nk - 1) & 1);
#else
  // ---- register-prefetch double-buffered pipeline --------------------------
  uint4 pa0, pa1, pb0, pb1;
  auto fetch_tile = [&](int k0) {
    pa0 = *(const uint4*)(gA0 + k0);
    pa1 = *(const uint4*)(gA1 + k0);
    pb0 = *(const uint4*)(gB0 + k0);
    pb1 = *(const uint4*)(gB1 + k0);
  };
  auto store_tile = [&](int b) {
    *(uint4*)(&As[b][lofs0]) = pa0;
    *(uint4*)(&As[b][lofs1]) = pa1;
    *(uint4*)(&Bs[b][lofs0]) = pb0;
    *(uint4*)(&Bs[b][lofs1]) = pb1;
  };
  fetch_tile(0);
  store_tile(0);
  __syncthreads();
  for (int kt = 0; kt < nk; ++kt) {
    if (kt + 1 < nk) fetch_tile((kt + 1) * BK);   // global loads in flight
    compute(kt & 1);                              // overlap with WMMA
    if (kt + 1 < nk) {
      __syncthreads();                            // readers of other buf done
      store_tile((kt + 1) & 1);
      __syncthreads();                            // visible for next iteration
    }
  }
#endif

  // --- epilogue: C/D layout: VGPR r, lanes0-15 -> M=r, lanes16-31 -> M=8+r
  const int rowoff = kg * 8;
#pragma unroll
  for (int j = 0; j < 4; ++j) {
    int col = tileN + wn * 64 + j * 16 + ml;
    float bv = bias[col];
#pragma unroll
    for (int i = 0; i < 2; ++i) {
#pragma unroll
      for (int r = 0; r < 8; ++r) {
        int grow = tileM + wm * 32 + i * 16 + rowoff + r;
        if (grow < M) {
          float v = acc[i][j][r] + bv;
          if (RELU) v = v > 0.f ? v : 0.f;
          C[(size_t)grow * Ncols + col] = (OutT)v;
        }
      }
    }
  }
}

// ---------------------------------------------------------------------------
extern "C" void kernel_launch(void* const* d_in, const int* in_sizes, int n_in,
                              void* d_out, int out_size, void* d_ws,
                              size_t ws_size, hipStream_t stream) {
  const float* x   = (const float*)d_in[0];
  const float* ew  = (const float*)d_in[1];
  const float* W1  = (const float*)d_in[2];
  const float* b1  = (const float*)d_in[3];
  const float* W2  = (const float*)d_in[4];
  const float* b2  = (const float*)d_in[5];
  const float* W3  = (const float*)d_in[6];
  const float* b3  = (const float*)d_in[7];
  const int*  esrc = (const int*)d_in[8];
  const int*  edst = (const int*)d_in[9];
  float* out = (float*)d_out;

  // workspace layout (256-B aligned slices)
  char*  ws  = (char*)d_ws;
  size_t off = 0;
  auto alloc = [&](size_t bytes) -> void* {
    void* p = ws + off;
    off = (off + bytes + 255) & ~(size_t)255;
    return p;
  };
  float* agg = (float*)alloc((size_t)NNODES * IN_DIM * sizeof(float));
  bf16*  Wt1 = (bf16*) alloc((size_t)CAT_DIM * H1_DIM * sizeof(bf16));
  bf16*  Wt2 = (bf16*) alloc((size_t)H1_DIM * H2_DIM * sizeof(bf16));
  bf16*  Wt3 = (bf16*) alloc((size_t)H2_DIM * OUT_DIM * sizeof(bf16));
  bf16*  hc  = (bf16*) alloc((size_t)NNODES * CAT_DIM * sizeof(bf16));
  bf16*  h1  = (bf16*) alloc((size_t)NNODES * H1_DIM * sizeof(bf16));
  bf16*  h2  = (bf16*) alloc((size_t)NNODES * H2_DIM * sizeof(bf16));

  // 1) agg = 0 ; scatter-add SpMM
  (void)hipMemsetAsync(agg, 0, (size_t)NNODES * IN_DIM * sizeof(float), stream);
  spmm_scatter<<<NEDGES, 128, 0, stream>>>(x, ew, esrc, edst, agg);

  // 2) weight transpose-convert to bf16 (N x K)
  {
    size_t n1 = (size_t)CAT_DIM * H1_DIM;
    size_t n2 = (size_t)H1_DIM * H2_DIM;
    size_t n3 = (size_t)H2_DIM * OUT_DIM;
    transpose_w_bf16<<<(unsigned)((n1 + 255) / 256), 256, 0, stream>>>(W1, Wt1, CAT_DIM, H1_DIM);
    transpose_w_bf16<<<(unsigned)((n2 + 255) / 256), 256, 0, stream>>>(W2, Wt2, H1_DIM, H2_DIM);
    transpose_w_bf16<<<(unsigned)((n3 + 255) / 256), 256, 0, stream>>>(W3, Wt3, H2_DIM, OUT_DIM);
  }

  // 3) hcat = bf16(concat(x, agg))
  {
    size_t tot = (size_t)NNODES * CAT_DIM;
    build_hcat<<<(unsigned)((tot + 255) / 256), 256, 0, stream>>>(x, agg, hc);
  }

  // 4) MLP: three WMMA GEMMs (M=50000 -> 391 row tiles of 128)
  const unsigned gy = (NNODES + 127) / 128;
  gemm_wmma_bf16<true, bf16><<<dim3(H1_DIM / 128, gy), 256, 0, stream>>>(
      hc, Wt1, b1, h1, NNODES, CAT_DIM, H1_DIM);
  gemm_wmma_bf16<true, bf16><<<dim3(H2_DIM / 128, gy), 256, 0, stream>>>(
      h1, Wt2, b2, h2, NNODES, H1_DIM, H2_DIM);
  gemm_wmma_bf16<false, float><<<dim3(OUT_DIM / 128, gy), 256, 0, stream>>>(
      h2, Wt3, b3, out, NNODES, H2_DIM, OUT_DIM);
}
